// GTModel_50208167690436
// MI455X (gfx1250) — compile-verified, matrix operand order checked by reading, not run on hardware
//
#include <hip/hip_runtime.h>
#include <hip/hip_bf16.h>

// ---------------------------------------------------------------------------
// Graph-transformer for MI455X (gfx1250, wave32).
// Dense QKV GEMMs -> v_wmma_f32_16x16x32_bf16 (bf16 inputs, f32 accum),
//   register-blocked (5 M-tiles/wave, B fragments held in VGPRs) with the
//   A tile staged via CDNA5 async global->LDS DMA (ASYNCcnt-tracked).
// Sparse SDDMM/softmax/SpMM -> VALU + float atomics (L2-resident gathers),
//   with global_prefetch warming L2 for the SpMM's random v[col] rows.
// ---------------------------------------------------------------------------

#define N_NODES 50000
#define N_EDGES 800000
#define HID     128
#define POS     16
#define LAYERS  8
#define HEADS   8
#define HDIM    16
#define OUTD    10
#define N_GRAPHS 128
#define MTILES  5                    // row tiles per block (80 rows)

typedef __bf16 bf16;
typedef __attribute__((ext_vector_type(16))) __bf16 v16bf;
typedef __attribute__((ext_vector_type(8)))  __bf16 v8bf;
typedef __attribute__((ext_vector_type(8)))  float  v8f;

// -------------------- float atomic max via int atomics ---------------------
__device__ inline void atomicMaxFloat(float* addr, float val) {
    if (val >= 0.0f) {
        atomicMax((int*)addr, __float_as_int(val));
    } else {
        atomicMin((unsigned int*)addr, __float_as_uint(val));
    }
}

// -------------------- h = pos_enc @ Wpos + bpos -----------------------------
__global__ __launch_bounds__(256) void pos_kernel(
    const float* __restrict__ pos_enc, const float* __restrict__ Wpos,
    const float* __restrict__ bpos, float* __restrict__ h, bf16* __restrict__ hbf)
{
    int idx = blockIdx.x * 256 + threadIdx.x;         // n*128 + c
    if (idx >= N_NODES * HID) return;
    int n = idx >> 7, c = idx & 127;
    float s = bpos[c];
    const float* pe = pos_enc + (size_t)n * POS;
#pragma unroll
    for (int p = 0; p < POS; ++p) s += pe[p] * Wpos[p * HID + c];
    h[idx] = s;
    hbf[idx] = (bf16)s;
}

// ---- pre-swizzle all 24 weight matrices into WMMA B-fragment layout --------
// wfrag[layer][which][nt][kk][lane][j]  (bf16), one-to-one with W elements.
__global__ __launch_bounds__(256) void wconv_kernel(
    const float* __restrict__ Wq, const float* __restrict__ Wk,
    const float* __restrict__ Wv, bf16* __restrict__ wfrag)
{
    int idx = blockIdx.x * 256 + threadIdx.x;
    if (idx >= LAYERS * 3 * HID * HID) return;
    int m3   = idx / 16384;          // layer*3 + which
    int rem  = idx % 16384;
    int nt   = rem / 2048;
    int rem2 = rem % 2048;
    int kk   = rem2 / 512;
    int rem3 = rem2 % 512;
    int lane = rem3 / 16;
    int j    = rem3 % 16;
    int n      = nt * 16 + (lane & 15);
    int klocal = ((lane >> 4) * 8) + (j & 7) + ((j >= 8) ? 16 : 0);
    int k      = kk * 32 + klocal;
    int layer  = m3 / 3, which = m3 % 3;
    const float* W = (which == 0) ? Wq : (which == 1) ? Wk : Wv;
    wfrag[idx] = (bf16)W[((size_t)layer * HID + k) * HID + n];
}

// -------------------- WMMA QKV GEMM -----------------------------------------
// grid = (625 super-row-tiles, 3 matrices). 256 threads = 8 waves; wave w owns
// output column tile w for all 5 row tiles of this block. B fragments are
// loaded once into VGPRs and reused across the 5 M-tiles -> 20 WMMAs/wave.
// The 80x128 bf16 A tile is staged with async global->LDS DMA (no VGPR trip).
__global__ __launch_bounds__(256) void qkv_gemm_kernel(
    const bf16* __restrict__ hbf, const bf16* __restrict__ wfrag_l,
    const float* __restrict__ bq, const float* __restrict__ bk,
    const float* __restrict__ bv,
    float* __restrict__ q, float* __restrict__ k, float* __restrict__ v)
{
    __shared__ bf16 Alds[MTILES * 16 * HID];          // 20 KB tile of h (bf16)
    const int blk   = blockIdx.x;                     // super row tile (80 rows)
    const int which = blockIdx.y;                     // 0=q,1=k,2=v
    const int tid   = threadIdx.x;

    // async global->LDS copy: 2560 x 16B across 256 lanes, ASYNCcnt-tracked.
    // LDS dest (wave-relative byte offset) in a VGPR, 64-bit global addr in a
    // VGPR pair, SADDR=off.  (ISA: GLOBAL_LOAD_ASYNC_TO_LDS_B128, GV mode)
    {
        unsigned laddr = (unsigned)(size_t)(&Alds[0]) + (unsigned)tid * 16u;
        const char* g = (const char*)(hbf + (size_t)blk * (MTILES * 16) * HID)
                        + (size_t)tid * 16;
#pragma unroll
        for (int i = 0; i < MTILES * 2; ++i) {
            asm volatile("global_load_async_to_lds_b128 %0, %1, off"
                         :: "v"(laddr + (unsigned)(i * 4096)),
                            "v"(g + (size_t)i * 4096)
                         : "memory");
        }
        asm volatile("s_wait_asynccnt 0x0" ::: "memory");
    }
    __syncthreads();

    const int wave = tid >> 5;                        // column tile 0..7
    const int lane = tid & 31;
    const int l15  = lane & 15;
    const int kgrp = (lane >> 4) * 8;

    const float* bias = (which == 0) ? bq : (which == 1) ? bk : bv;
    const float bval = bias[wave * 16 + l15];         // C column = lane&15
    const float scale = (which == 0) ? 0.25f : 1.0f;  // HEAD_DIM^-0.5
    float* out = (which == 0) ? q : (which == 1) ? k : v;

    // B fragments: one contiguous 32B load per k-step, kept in registers
    const bf16* wbase = wfrag_l + (((size_t)which * 8 + wave) * 4) * 512;
    v16bf bfrag[4];
#pragma unroll
    for (int kk = 0; kk < 4; ++kk)
        bfrag[kk] = *(const v16bf*)(wbase + kk * 512 + lane * 16);

#pragma unroll
    for (int mt = 0; mt < MTILES; ++mt) {
        v8f c;
#pragma unroll
        for (int r = 0; r < 8; ++r) c[r] = bval;
#pragma unroll
        for (int kk = 0; kk < 4; ++kk) {
            // A fragment per ISA 16-bit 16x32 layout: two 16B LDS reads
            const bf16* ap = &Alds[(mt * 16 + l15) * HID + kk * 32 + kgrp];
            v8bf alo = *(const v8bf*)(ap);
            v8bf ahi = *(const v8bf*)(ap + 16);
            v16bf a;
#pragma unroll
            for (int j = 0; j < 8; ++j) { a[j] = alo[j]; a[8 + j] = ahi[j]; }
            c = __builtin_amdgcn_wmma_f32_16x16x32_bf16(
                    false, a, false, bfrag[kk], (short)0, c, false, false);
        }
#pragma unroll
        for (int r = 0; r < 8; ++r) {
            int m = mt * 16 + r + 8 * (lane >> 4);    // C layout, VGPR r
            out[((size_t)blk * (MTILES * 16) + m) * HID + wave * 16 + l15]
                = c[r] * scale;
        }
    }
}

// -------------------- per-layer init: hnext=0, m=-inf, denom=0 --------------
__global__ __launch_bounds__(256) void init_kernel(
    float* __restrict__ hnext, float* __restrict__ mbuf, float* __restrict__ denom)
{
    int idx = blockIdx.x * 256 + threadIdx.x;
    if (idx < N_NODES * HID) hnext[idx] = 0.0f;
    if (idx < N_NODES * HEADS) { mbuf[idx] = -__builtin_inff(); denom[idx] = 0.0f; }
}

// -------------------- SDDMM: score[e,h] = q[row]·k[col], seg-max ------------
// Also issues global_prefetch for v[col] rows to warm L2 for aggregation.
__global__ __launch_bounds__(256) void sddmm_kernel(
    const int* __restrict__ row, const int* __restrict__ col,
    const float* __restrict__ q, const float* __restrict__ k,
    const float* __restrict__ v,
    float* __restrict__ scores, float* __restrict__ mbuf)
{
    int idx = blockIdx.x * 256 + threadIdx.x;         // e*8 + h
    if (idx >= N_EDGES * HEADS) return;
    int e = idx >> 3, hh = idx & 7;
    int r = row[e], c = col[e];
    const float* qp = q + (size_t)r * HID + hh * HDIM;
    const float* kp = k + (size_t)c * HID + hh * HDIM;
    __builtin_prefetch(v + (size_t)c * HID + hh * HDIM, 0, 1);  // warm L2 for SpMM
    float s = 0.0f;
#pragma unroll
    for (int d = 0; d < HDIM; ++d) s += qp[d] * kp[d];
    scores[idx] = s;
    atomicMaxFloat(&mbuf[r * HEADS + hh], s);
}

// -------------------- exp(score - m[row]) and segment-sum denom -------------
__global__ __launch_bounds__(256) void expdenom_kernel(
    const int* __restrict__ row, float* __restrict__ scores,
    const float* __restrict__ mbuf, float* __restrict__ denom)
{
    int idx = blockIdx.x * 256 + threadIdx.x;
    if (idx >= N_EDGES * HEADS) return;
    int e = idx >> 3, hh = idx & 7;
    int r = row[e];
    float ex = __expf(scores[idx] - mbuf[r * HEADS + hh]);
    scores[idx] = ex;
    atomicAdd(&denom[r * HEADS + hh], ex);
}

// -------------------- SpMM: hnext[row] += attn * v[col] ---------------------
__global__ __launch_bounds__(256) void aggregate_kernel(
    const int* __restrict__ row, const int* __restrict__ col,
    const float* __restrict__ scores, const float* __restrict__ denom,
    const float* __restrict__ v, float* __restrict__ hnext)
{
    int idx = blockIdx.x * 256 + threadIdx.x;
    if (idx >= N_EDGES * HEADS) return;
    int e = idx >> 3, hh = idx & 7;
    int r = row[e], c = col[e];
    float attn = scores[idx] / denom[r * HEADS + hh];
    const float* vp = v + (size_t)c * HID + hh * HDIM;
    float* op = hnext + (size_t)r * HID + hh * HDIM;
#pragma unroll
    for (int d = 0; d < HDIM; ++d) atomicAdd(&op[d], attn * vp[d]);
}

// -------------------- h <- hnext (fp32 + bf16 copies) -----------------------
__global__ __launch_bounds__(256) void finish_kernel(
    const float* __restrict__ hnext, float* __restrict__ h, bf16* __restrict__ hbf)
{
    int idx = blockIdx.x * 256 + threadIdx.x;
    if (idx >= N_NODES * HID) return;
    float x = hnext[idx];
    h[idx] = x;
    hbf[idx] = (bf16)x;
}

// -------------------- pooling -----------------------------------------------
__global__ __launch_bounds__(256) void zero_pooled_kernel(float* __restrict__ pooled)
{
    int idx = blockIdx.x * 256 + threadIdx.x;
    if (idx < N_GRAPHS * HID) pooled[idx] = 0.0f;
}

__global__ __launch_bounds__(256) void pool_kernel(
    const float* __restrict__ h, const int* __restrict__ graph_ids,
    float* __restrict__ pooled)
{
    int idx = blockIdx.x * 256 + threadIdx.x;
    if (idx >= N_NODES * HID) return;
    int n = idx >> 7, c = idx & 127;
    atomicAdd(&pooled[graph_ids[n] * HID + c], h[idx]);
}

// -------------------- prediction head ---------------------------------------
__global__ __launch_bounds__(256) void pred_kernel(
    const float* __restrict__ pooled, const float* __restrict__ Wpred,
    const float* __restrict__ bpred, float* __restrict__ out)
{
    int idx = blockIdx.x * 256 + threadIdx.x;         // g*10 + o
    if (idx >= N_GRAPHS * OUTD) return;
    int g = idx / OUTD, o = idx % OUTD;
    float s = bpred[o];
    const float* pp = pooled + (size_t)g * HID;
#pragma unroll 8
    for (int c = 0; c < HID; ++c) s += pp[c] * Wpred[c * OUTD + o];
    out[idx] = s;
}

// ===========================================================================
extern "C" void kernel_launch(void* const* d_in, const int* in_sizes, int n_in,
                              void* d_out, int out_size, void* d_ws, size_t ws_size,
                              hipStream_t stream) {
    (void)in_sizes; (void)n_in; (void)out_size; (void)ws_size;
    const float* pos_enc   = (const float*)d_in[1];
    const int*   edge_row  = (const int*)d_in[2];
    const int*   edge_col  = (const int*)d_in[3];
    const int*   graph_ids = (const int*)d_in[4];
    const float* Wpos      = (const float*)d_in[5];
    const float* bpos      = (const float*)d_in[6];
    const float* Wq        = (const float*)d_in[7];
    const float* bq        = (const float*)d_in[8];
    const float* Wk        = (const float*)d_in[9];
    const float* bk        = (const float*)d_in[10];
    const float* Wv        = (const float*)d_in[11];
    const float* bv        = (const float*)d_in[12];
    const float* Wpred     = (const float*)d_in[13];
    const float* bpred     = (const float*)d_in[14];
    float* out = (float*)d_out;

    // ---- workspace carve-up (all sizes are multiples of 256 bytes) ----
    char* ws = (char*)d_ws;
    size_t off = 0;
    auto take = [&](size_t bytes) { char* p = ws + off; off += bytes; return p; };
    float* h      = (float*)take((size_t)N_NODES * HID * 4);
    float* hnext  = (float*)take((size_t)N_NODES * HID * 4);
    float* q      = (float*)take((size_t)N_NODES * HID * 4);
    float* k      = (float*)take((size_t)N_NODES * HID * 4);
    float* v      = (float*)take((size_t)N_NODES * HID * 4);
    float* scores = (float*)take((size_t)N_EDGES * HEADS * 4);
    float* mbuf   = (float*)take((size_t)N_NODES * HEADS * 4);
    float* denom  = (float*)take((size_t)N_NODES * HEADS * 4);
    bf16*  hbf    = (bf16*)take((size_t)N_NODES * HID * 2);
    bf16*  wfrag  = (bf16*)take((size_t)LAYERS * 3 * HID * HID * 2);
    float* pooled = (float*)take((size_t)N_GRAPHS * HID * 4);

    const int NC_BLOCKS = (N_NODES * HID) / 256;      // 25000
    const int EH_BLOCKS = (N_EDGES * HEADS) / 256;    // 25000

    pos_kernel<<<NC_BLOCKS, 256, 0, stream>>>(pos_enc, Wpos, bpos, h, hbf);
    wconv_kernel<<<(LAYERS * 3 * HID * HID) / 256, 256, 0, stream>>>(Wq, Wk, Wv, wfrag);

    for (int layer = 0; layer < LAYERS; ++layer) {
        const bf16* wfrag_l = wfrag + (size_t)layer * 3 * HID * HID;
        qkv_gemm_kernel<<<dim3(N_NODES / (MTILES * 16), 3), 256, 0, stream>>>(
            hbf, wfrag_l, bq + layer * HID, bk + layer * HID, bv + layer * HID,
            q, k, v);
        init_kernel<<<NC_BLOCKS, 256, 0, stream>>>(hnext, mbuf, denom);
        sddmm_kernel<<<EH_BLOCKS, 256, 0, stream>>>(edge_row, edge_col, q, k, v, scores, mbuf);
        expdenom_kernel<<<EH_BLOCKS, 256, 0, stream>>>(edge_row, scores, mbuf, denom);
        aggregate_kernel<<<EH_BLOCKS, 256, 0, stream>>>(edge_row, edge_col, scores, denom, v, hnext);
        finish_kernel<<<NC_BLOCKS, 256, 0, stream>>>(hnext, h, hbf);
    }

    zero_pooled_kernel<<<(N_GRAPHS * HID) / 256, 256, 0, stream>>>(pooled);
    pool_kernel<<<NC_BLOCKS, 256, 0, stream>>>(h, graph_ids, pooled);
    pred_kernel<<<(N_GRAPHS * OUTD + 255) / 256, 256, 0, stream>>>(pooled, Wpred, bpred, out);
}
